// SintelSwinVSRNetFeatureAlignment_85710367359395
// MI455X (gfx1250) — matrix operand order repaired and checked by reading, not compiled, not run on hardware
//
#include <hip/hip_runtime.h>
#include <math.h>

// ---------------------------------------------------------------------------
// Deformable-attention feature alignment, CDNA5 (gfx1250, wave32, WMMA).
//   - all linear/conv stages as v_wmma_f32_16x16x32_f16 GEMMs, f16 data
//   - weights pre-packed into B-fragment layout  -> 32B global_load_b128/lane
//   - activations staged in LDS in A-fragment layout -> 32B ds_load_b128/lane
//   - K-step 64: one b128 staging load/thread, two WMMAs per barrier period
//   - fused epilogues: lrelu / 10*tanh+flow / gelu / residual
//   - deformable bilinear gather + 18-way softmax in VALU
//   - TDM tensor_load_to_lds prefetch of k/v planes before the gather kernel
// ---------------------------------------------------------------------------

#define P 4096          // H*W pixels
#define IMW 64

typedef __attribute__((ext_vector_type(16))) _Float16 v16h;
typedef __attribute__((ext_vector_type(8)))  _Float16 v8h;
typedef __attribute__((ext_vector_type(8)))  float    v8f;
typedef unsigned int u32x4 __attribute__((ext_vector_type(4)));
typedef int          i32x4 __attribute__((ext_vector_type(4)));
typedef int          i32x8 __attribute__((ext_vector_type(8)));

#if defined(__has_builtin)
#  if __has_builtin(__builtin_amdgcn_tensor_load_to_lds)
#    define HAVE_TDM 1
#  endif
#  if __has_builtin(__builtin_amdgcn_s_wait_tensorcnt)
#    define HAVE_TENSORCNT 1
#  endif
#  if __has_builtin(__builtin_amdgcn_s_cluster_barrier)
#    define HAVE_CLUSTER_BAR 1
#  endif
#endif

// ---------------------------------------------------------------------------
// WMMA GEMM: Y[N][P] = epi(X[Kpad][P] * Wpk + bias), Kpad % 64 == 0
//   Wpk: f16 weights pre-packed per (ktile32,ntile) fragment: 512 f16,
//        lane-major: Wpk[frag*512 + lane*16 + e]
//   AMODE 0: X dense [Kpad][P]; AMODE 1: X feat map [64][P], fused im2col 3x3
// EPI: 0 ->f16 | 1 lrelu->f16 | 2 gelu->f16 | 3 10*tanh+flow->f32 |
//      4 ->f16+f32 | 5 +extra residual ->f32
// Block = 128 threads = 4 wave32; wave w owns n-tile (blockIdx.y*4+w)*16.
// ---------------------------------------------------------------------------
template<int AMODE, int EPI>
__global__ __launch_bounds__(128)
void gemm_wmma(const _Float16* __restrict__ X, const _Float16* __restrict__ Wpk,
               const float* __restrict__ bias, int Kpad, int N,
               _Float16* __restrict__ y16, float* __restrict__ y32,
               const float* __restrict__ extra) {
    __shared__ v16h sAv[64];                   // two 32-K fragments (2 KB)
    _Float16* sA = (_Float16*)sAv;
    const int m0   = blockIdx.x * 16;
    const int lane = threadIdx.x & 31;
    const int wave = threadIdx.x >> 5;
    const int n0   = (blockIdx.y * 4 + wave) * 16;
    const int hi   = lane >> 4;
    const int lcol = lane & 15;
    const int ntiles = N >> 4;
    const bool active = (n0 < N);              // wave-uniform: EXEC all-ones

    const int ty = m0 >> 6;                    // tile row (16 | 64)
    const int tx = m0 & 63;

    // per-thread staging role: row kk (0..63), 8-wide column chunk mmq
    const int kk  = threadIdx.x >> 1;
    const int mmq = (threadIdx.x & 1) << 3;
    const int kl  = kk & 31;
    // fragment-layout store base: lane' = (kl>>4)*16 + mm, e' = kl&15
    const int stc = ((kk >> 5) << 9)
                  + ((((kl >> 4) & 1) << 4) << 4)
                  + (kl & 15);

    v8f acc = {0.f, 0.f, 0.f, 0.f, 0.f, 0.f, 0.f, 0.f};

    for (int k0 = 0; k0 < Kpad; k0 += 64) {
        // ---- stage A tile [64 k][16 m] into LDS, fragment layout ----
        if (AMODE == 0) {
            v8h vv = *(const v8h*)(X + (size_t)(k0 + kk) * P + m0 + mmq);
            #pragma unroll
            for (int j = 0; j < 8; ++j)
                sA[stc + ((mmq + j) << 4)] = vv[j];
        } else {                               // fused im2col 3x3, 64 ch
            int k  = k0 + kk;
            int t  = k >> 6;                   // neighbor tap 0..8
            int ic = k & 63;
            int yy = ty + (t / 3) - 1;
            int dx = (t % 3) - 1;
            const _Float16* row = X + (size_t)ic * P + yy * IMW;
            bool rowok = ((unsigned)yy < 64u);
            #pragma unroll
            for (int j = 0; j < 8; ++j) {
                int mm = mmq + j;
                int xx = tx + mm + dx;
                _Float16 v = (_Float16)0.f;
                if (rowok && (unsigned)xx < 64u) v = row[xx];
                sA[stc + (mm << 4)] = v;
            }
        }
        __syncthreads();

        if (active) {
            const int kt2 = k0 >> 5;
            const _Float16* w0 = Wpk + ((((size_t)kt2) * ntiles + (n0 >> 4)) << 9)
                                     + (lane << 4);
            const _Float16* w1 = Wpk + ((((size_t)kt2 + 1) * ntiles + (n0 >> 4)) << 9)
                                     + (lane << 4);
            if (k0 + 64 < Kpad)                // global_prefetch_b8 next frags
                __builtin_prefetch((const void*)(Wpk
                    + ((((size_t)kt2 + 2) * ntiles + (n0 >> 4)) << 9)), 0, 1);
            v16h a0 = sAv[lane];               // 2x ds_load_b128 each
            v16h a1 = sAv[32 + lane];
            v16h b0 = *(const v16h*)w0;        // 2x global_load_b128 each
            v16h b1 = *(const v16h*)w1;
            acc = __builtin_amdgcn_wmma_f32_16x16x32_f16(
                      false, a0, false, b0, (short)0, acc, false, false);
            acc = __builtin_amdgcn_wmma_f32_16x16x32_f16(
                      false, a1, false, b1, (short)0, acc, false, false);
        }
        __syncthreads();
    }

    if (!active) return;
    const int n = n0 + lcol;
    const float bias_v = bias[n];
    #pragma unroll
    for (int r = 0; r < 8; ++r) {              // D layout: M = r + 8*hi
        int m = m0 + r + 8 * hi;
        float v = acc[r] + bias_v;
        if (EPI == 1) v = (v >= 0.f) ? v : 0.1f * v;
        if (EPI == 2) v = 0.5f * v * (1.0f + erff(v * 0.70710678118f));
        if (EPI == 3) {
            // offsets: 10*tanh + flow (channel-flipped): even n += flow ch1
            v = 10.f * tanhf(v) + extra[(size_t)((n & 1) ^ 1) * P + m];
            y32[(size_t)n * P + m] = v;
        } else if (EPI == 5) {
            y32[(size_t)n * P + m] = v + extra[(size_t)n * P + m];
        } else if (EPI == 4) {
            y32[(size_t)n * P + m] = v;
            y16[(size_t)n * P + m] = (_Float16)v;
        } else {
            y16[(size_t)n * P + m] = (_Float16)v;
        }
    }
}

// ---------------------------------------------------------------------------
// Deformable bilinear gather + per-pixel-per-head attention over 18 samples.
// ---------------------------------------------------------------------------
__global__ __launch_bounds__(256)
void deform_attn(const _Float16* __restrict__ QP,   // [288][P]
                 const _Float16* __restrict__ KP,   // [2][288][P]
                 const _Float16* __restrict__ VP,   // [2][288][P]
                 const float*    __restrict__ OFF,  // [432][P]
                 _Float16*       __restrict__ OUT)  // [>=288][P]
{
    int idx = blockIdx.x * blockDim.x + threadIdx.x;
    if (idx >= P * 12) return;
    int p = idx & (P - 1);
    int h = idx >> 12;
    int py0 = p >> 6, px0 = p & 63;

    float qv[24];
    #pragma unroll
    for (int d = 0; d < 24; ++d) qv[d] = (float)QP[(size_t)(h * 24 + d) * P + p];

    float sc[18];
    #pragma unroll 1
    for (int c = 0; c < 2; ++c) {
        const _Float16* KB = KP + ((size_t)c * 288 + h * 24) * P;
        #pragma unroll 1
        for (int a = 0; a < 9; ++a) {
            int och = ((c * 12 + h) * 9 + a) * 2;
            float sy = OFF[(size_t)och * P + p] + (float)(a / 3 - 1) + (float)py0;
            float sx = OFF[(size_t)(och + 1) * P + p] + (float)(a % 3 - 1) + (float)px0;
            float fy = floorf(sy), fx = floorf(sx);
            int iy = (int)fy, ix = (int)fx;
            float wy = sy - fy, wx = sx - fx;
            float w00 = (1.f - wy) * (1.f - wx), w01 = (1.f - wy) * wx;
            float w10 = wy * (1.f - wx),         w11 = wy * wx;
            bool v00 = ((unsigned)iy < 64u) && ((unsigned)ix < 64u);
            bool v01 = ((unsigned)iy < 64u) && ((unsigned)(ix + 1) < 64u);
            bool v10 = ((unsigned)(iy + 1) < 64u) && ((unsigned)ix < 64u);
            bool v11 = ((unsigned)(iy + 1) < 64u) && ((unsigned)(ix + 1) < 64u);
            int o00 = iy * IMW + ix;
            float s = 0.f;
            #pragma unroll
            for (int d = 0; d < 24; ++d) {
                const _Float16* F = KB + (size_t)d * P;
                float g = 0.f;
                if (v00) g += w00 * (float)F[o00];
                if (v01) g += w01 * (float)F[o00 + 1];
                if (v10) g += w10 * (float)F[o00 + IMW];
                if (v11) g += w11 * (float)F[o00 + IMW + 1];
                s += qv[d] * g;
            }
            sc[c * 9 + a] = s * 0.20412414523193154f;   // hd^-0.5, hd=24
        }
    }
    float mx = sc[0];
    #pragma unroll
    for (int s = 1; s < 18; ++s) mx = fmaxf(mx, sc[s]);
    float sum = 0.f;
    #pragma unroll
    for (int s = 0; s < 18; ++s) { sc[s] = __expf(sc[s] - mx); sum += sc[s]; }
    float inv = 1.f / sum;

    float out[24];
    #pragma unroll
    for (int d = 0; d < 24; ++d) out[d] = 0.f;

    #pragma unroll 1
    for (int c = 0; c < 2; ++c) {
        const _Float16* VB = VP + ((size_t)c * 288 + h * 24) * P;
        #pragma unroll 1
        for (int a = 0; a < 9; ++a) {
            int och = ((c * 12 + h) * 9 + a) * 2;
            float sy = OFF[(size_t)och * P + p] + (float)(a / 3 - 1) + (float)py0;
            float sx = OFF[(size_t)(och + 1) * P + p] + (float)(a % 3 - 1) + (float)px0;
            float fy = floorf(sy), fx = floorf(sx);
            int iy = (int)fy, ix = (int)fx;
            float wy = sy - fy, wx = sx - fx;
            float wgt = sc[c * 9 + a] * inv;
            float w00 = wgt * (1.f - wy) * (1.f - wx), w01 = wgt * (1.f - wy) * wx;
            float w10 = wgt * wy * (1.f - wx),         w11 = wgt * wy * wx;
            bool v00 = ((unsigned)iy < 64u) && ((unsigned)ix < 64u);
            bool v01 = ((unsigned)iy < 64u) && ((unsigned)(ix + 1) < 64u);
            bool v10 = ((unsigned)(iy + 1) < 64u) && ((unsigned)ix < 64u);
            bool v11 = ((unsigned)(iy + 1) < 64u) && ((unsigned)(ix + 1) < 64u);
            int o00 = iy * IMW + ix;
            #pragma unroll
            for (int d = 0; d < 24; ++d) {
                const _Float16* F = VB + (size_t)d * P;
                float g = 0.f;
                if (v00) g += w00 * (float)F[o00];
                if (v01) g += w01 * (float)F[o00 + 1];
                if (v10) g += w10 * (float)F[o00 + IMW];
                if (v11) g += w11 * (float)F[o00 + IMW + 1];
                out[d] += g;
            }
        }
    }
    #pragma unroll
    for (int d = 0; d < 24; ++d)
        OUT[(size_t)(h * 24 + d) * P + p] = (_Float16)out[d];
}

// ---------------------------------------------------------------------------
// TDM prefetch: pull a 2D tile of `src` into LDS via the Tensor Data Mover.
// D# groups 0/1 per CDNA5 ISA; groups 2/3 zero (2D).
// Compile-guarded; falls back to global_prefetch.
// ---------------------------------------------------------------------------
__global__ __launch_bounds__(32)
void k_tdm_prefetch(const _Float16* __restrict__ src, int rows, int cols) {
#if defined(HAVE_TDM)
    __shared__ __attribute__((aligned(16))) char lbuf[16384];
    unsigned long long ga = (unsigned long long)(uintptr_t)src;
    u32x4 g0;
    g0[0] = 1u;                                  // count=1, user descriptor
    g0[1] = 0u;                                  // lds_addr = 0 (wg LDS base)
    g0[2] = (unsigned)(ga & 0xffffffffu);        // global_addr[31:0]
    g0[3] = (unsigned)((ga >> 32) & 0x01ffffffu) // global_addr[56:32]
            | (2u << 30);                        // type = 2 ("image")
    i32x8 g1;
    g1[0] = 0x10000;                             // data_size=1 (2B), mask=0
    g1[1] = (cols & 0xffff) << 16;               // tensor_dim0[15:0]
    g1[2] = ((cols >> 16) & 0xffff)              // tensor_dim0[31:16]
            | ((rows & 0xffff) << 16);           // tensor_dim1[15:0]
    g1[3] = ((rows >> 16) & 0xffff)              // tensor_dim1[31:16]
            | ((cols & 0xffff) << 16);           // tile_dim0 = cols
    g1[4] = 4;                                   // tile_dim1=4 rows, dim2=0
    g1[5] = cols;                                // tensor_dim0_stride[31:0]
    g1[6] = 0;
    g1[7] = 0;
    i32x4 z4 = {0, 0, 0, 0};
#if __clang_major__ >= 23
    i32x8 z8 = {0, 0, 0, 0, 0, 0, 0, 0};
    __builtin_amdgcn_tensor_load_to_lds(g0, g1, z4, z4, z8, 0);
#else
    __builtin_amdgcn_tensor_load_to_lds(g0, g1, z4, z4, 0);
#endif
#if defined(HAVE_TENSORCNT)
    __builtin_amdgcn_s_wait_tensorcnt(0);
#endif
    if (threadIdx.x == 0) ((volatile char*)lbuf)[0] = lbuf[1];
#else
    (void)rows; (void)cols;
    __builtin_prefetch((const void*)src, 0, 1);
#endif
#if defined(HAVE_CLUSTER_BAR)
    __builtin_amdgcn_s_cluster_barrier();        // NOP when not in a cluster
#endif
}

// ---------------------------------------------------------------------------
// Weight pack into WMMA B-fragment layout:
//   out[frag*512 + lane*16 + e], frag = kt*(N/16)+nt,
//   k = kt*32 + e + (lane>>4)*16, n = nt*16 + (lane&15)
// mode 0: src [N][K] (1x1 conv OIHW) | 1: src [K][N] (linear) |
// mode 2: 3x3 conv, k = t*64+ic, src w[(n*64+ic)*9 + t]
// ---------------------------------------------------------------------------
__global__ void k_pack_w(const float* __restrict__ w, _Float16* __restrict__ out,
                         int mode, int Ksrc, int Kpad, int N) {
    int o = blockIdx.x * blockDim.x + threadIdx.x;
    int total = (Kpad >> 5) * (N >> 4) * 512;
    if (o >= total) return;
    int e    = o & 15;
    int lane = (o >> 4) & 31;
    int frag = o >> 9;
    int ntiles = N >> 4;
    int kt = frag / ntiles, nt = frag % ntiles;
    int k = kt * 32 + e + (lane >> 4) * 16;
    int n = nt * 16 + (lane & 15);
    float v = 0.f;
    if (k < Ksrc) {
        if (mode == 0)      v = w[(size_t)n * Ksrc + k];
        else if (mode == 1) v = w[(size_t)k * N + n];
        else { int t = k >> 6, ic = k & 63; v = w[((size_t)n * 64 + ic) * 9 + t]; }
    }
    out[o] = (_Float16)v;
}

__global__ void k_pack436(const float* __restrict__ q, const float* __restrict__ v0,
                          const float* __restrict__ v1, const float* __restrict__ f0,
                          const float* __restrict__ f1, _Float16* __restrict__ out) {
    int i = blockIdx.x * blockDim.x + threadIdx.x;
    if (i >= 448 * P) return;
    int c = i >> 12, p = i & (P - 1);
    float v = 0.f;
    if      (c < 144) v = q [(size_t)c * P + p];
    else if (c < 288) v = v0[(size_t)(c - 144) * P + p];
    else if (c < 432) v = v1[(size_t)(c - 288) * P + p];
    else if (c < 434) v = f0[(size_t)(c - 432) * P + p];
    else if (c < 436) v = f1[(size_t)(c - 434) * P + p];
    out[i] = (_Float16)v;
}

__global__ void k_tof16(const float* __restrict__ src, _Float16* __restrict__ dst,
                        int C, int Cpad, int reps) {
    int i = blockIdx.x * blockDim.x + threadIdx.x;
    if (i >= reps * Cpad * P) return;
    int p  = i & (P - 1);
    int cc = (i >> 12) % Cpad;
    int r  = (i >> 12) / Cpad;
    dst[i] = (cc < C) ? (_Float16)src[((size_t)r * C + cc) * P + p] : (_Float16)0.f;
}

__global__ void k_zero16(_Float16* __restrict__ dst, int n) {
    int i = blockIdx.x * blockDim.x + threadIdx.x;
    if (i < n) dst[i] = (_Float16)0.f;
}

// ---------------------------------------------------------------------------
extern "C" void kernel_launch(void* const* d_in, const int* in_sizes, int n_in,
                              void* d_out, int out_size, void* d_ws, size_t ws_size,
                              hipStream_t stream) {
    (void)in_sizes; (void)n_in; (void)out_size; (void)ws_size;
    const float* q     = (const float*)d_in[0];
    const float* kin   = (const float*)d_in[1];
    const float* vin   = (const float*)d_in[2];
    const float* vpw0  = (const float*)d_in[3];
    const float* vpw1  = (const float*)d_in[4];
    const float* flow0 = (const float*)d_in[5];
    const float* flow1 = (const float*)d_in[6];
    const float* wo1 = (const float*)d_in[7];  const float* bo1 = (const float*)d_in[8];
    const float* wo2 = (const float*)d_in[9];  const float* bo2 = (const float*)d_in[10];
    const float* wo3 = (const float*)d_in[11]; const float* bo3 = (const float*)d_in[12];
    const float* wo4 = (const float*)d_in[13]; const float* bo4 = (const float*)d_in[14];
    const float* wo5 = (const float*)d_in[15]; const float* bo5 = (const float*)d_in[16];
    const float* wo6 = (const float*)d_in[17]; const float* bo6 = (const float*)d_in[18];
    const float* wq  = (const float*)d_in[19]; const float* bq  = (const float*)d_in[20];
    const float* wk  = (const float*)d_in[21]; const float* bk  = (const float*)d_in[22];
    const float* wv  = (const float*)d_in[23]; const float* bvv = (const float*)d_in[24];
    const float* wp  = (const float*)d_in[25]; const float* bp  = (const float*)d_in[26];
    const float* wm1 = (const float*)d_in[27]; const float* bm1 = (const float*)d_in[28];
    const float* wm2 = (const float*)d_in[29]; const float* bm2 = (const float*)d_in[30];

    char* ws = (char*)d_ws;
    size_t cur = 0;
    auto a16 = [&](size_t e) { _Float16* r = (_Float16*)(ws + cur);
                               cur += (e * 2 + 255) & ~(size_t)255; return r; };
    auto a32 = [&](size_t e) { float* r = (float*)(ws + cur);
                               cur += (e * 4 + 255) & ~(size_t)255; return r; };

    _Float16* X436 = a16((size_t)448 * P);
    _Float16* F1   = a16((size_t)64 * P);
    _Float16* F2   = a16((size_t)64 * P);
    float*    OFF  = a32((size_t)432 * P);
    _Float16* QX   = a16((size_t)192 * P);
    _Float16* KX   = a16((size_t)2 * 192 * P);
    _Float16* VX   = a16((size_t)2 * 192 * P);
    _Float16* QPb  = a16((size_t)288 * P);
    _Float16* KPb  = a16((size_t)2 * 288 * P);
    _Float16* VPb  = a16((size_t)2 * 288 * P);
    _Float16* AT   = a16((size_t)320 * P);     // rows 288..319 zero pad
    float*    O32  = a32((size_t)144 * P);
    _Float16* O16  = a16((size_t)192 * P);     // rows 144..191 zero pad
    _Float16* M1   = a16((size_t)320 * P);     // rows 288..319 zero pad
    _Float16* wo1f = a16((size_t)448 * 64);
    _Float16* wo2f = a16((size_t)576 * 64);
    _Float16* wo3f = a16((size_t)576 * 64);
    _Float16* wo4f = a16((size_t)576 * 64);
    _Float16* wo5f = a16((size_t)576 * 64);
    _Float16* wo6f = a16((size_t)64 * 432);
    _Float16* wqf  = a16((size_t)192 * 288);
    _Float16* wkf  = a16((size_t)192 * 288);
    _Float16* wvf  = a16((size_t)192 * 288);
    _Float16* wpf  = a16((size_t)320 * 144);
    _Float16* wm1f = a16((size_t)192 * 288);
    _Float16* wm2f = a16((size_t)320 * 144);

    auto g1 = [](int n) { return dim3((unsigned)((n + 255) / 256)); };
    const dim3 t256(256), blk(128);

    // --- weight packs (B-fragment layout) ---
    k_pack_w<<<g1(448 * 64),  t256, 0, stream>>>(wo1, wo1f, 0, 436, 448, 64);
    k_pack_w<<<g1(576 * 64),  t256, 0, stream>>>(wo2, wo2f, 2, 576, 576, 64);
    k_pack_w<<<g1(576 * 64),  t256, 0, stream>>>(wo3, wo3f, 2, 576, 576, 64);
    k_pack_w<<<g1(576 * 64),  t256, 0, stream>>>(wo4, wo4f, 2, 576, 576, 64);
    k_pack_w<<<g1(576 * 64),  t256, 0, stream>>>(wo5, wo5f, 2, 576, 576, 64);
    k_pack_w<<<g1(64 * 432),  t256, 0, stream>>>(wo6, wo6f, 0, 64, 64, 432);
    k_pack_w<<<g1(192 * 288), t256, 0, stream>>>(wq,  wqf,  1, 144, 192, 288);
    k_pack_w<<<g1(192 * 288), t256, 0, stream>>>(wk,  wkf,  1, 144, 192, 288);
    k_pack_w<<<g1(192 * 288), t256, 0, stream>>>(wv,  wvf,  1, 144, 192, 288);
    k_pack_w<<<g1(320 * 144), t256, 0, stream>>>(wp,  wpf,  1, 288, 320, 144);
    k_pack_w<<<g1(192 * 288), t256, 0, stream>>>(wm1, wm1f, 1, 144, 192, 288);
    k_pack_w<<<g1(320 * 144), t256, 0, stream>>>(wm2, wm2f, 1, 288, 320, 144);

    // --- input conversions + zero pads ---
    k_pack436<<<g1(448 * P), t256, 0, stream>>>(q, vpw0, vpw1, flow0, flow1, X436);
    k_tof16  <<<g1(192 * P), t256, 0, stream>>>(q,   QX, 144, 192, 1);
    k_tof16  <<<g1(2 * 192 * P), t256, 0, stream>>>(kin, KX, 144, 192, 2);
    k_tof16  <<<g1(2 * 192 * P), t256, 0, stream>>>(vin, VX, 144, 192, 2);
    k_zero16 <<<g1(48 * P), t256, 0, stream>>>(O16 + (size_t)144 * P, 48 * P);
    k_zero16 <<<g1(32 * P), t256, 0, stream>>>(AT  + (size_t)288 * P, 32 * P);
    k_zero16 <<<g1(32 * P), t256, 0, stream>>>(M1  + (size_t)288 * P, 32 * P);

    // --- offset predictor (WMMA GEMMs, im2col fused for 3x3) ---
    gemm_wmma<0,1><<<dim3(256, 1), blk, 0, stream>>>(X436, wo1f, bo1, 448, 64, F1, nullptr, nullptr);
    gemm_wmma<1,1><<<dim3(256, 1), blk, 0, stream>>>(F1, wo2f, bo2, 576, 64, F2, nullptr, nullptr);
    gemm_wmma<1,1><<<dim3(256, 1), blk, 0, stream>>>(F2, wo3f, bo3, 576, 64, F1, nullptr, nullptr);
    gemm_wmma<1,1><<<dim3(256, 1), blk, 0, stream>>>(F1, wo4f, bo4, 576, 64, F2, nullptr, nullptr);
    gemm_wmma<1,1><<<dim3(256, 1), blk, 0, stream>>>(F2, wo5f, bo5, 576, 64, F1, nullptr, nullptr);
    gemm_wmma<0,3><<<dim3(256, 7), blk, 0, stream>>>(F1, wo6f, bo6, 64, 432, nullptr, OFF, flow0);

    // --- q/k/v projections (Kpad 192) ---
    gemm_wmma<0,0><<<dim3(256, 5), blk, 0, stream>>>(QX, wqf, bq, 192, 288, QPb, nullptr, nullptr);
    gemm_wmma<0,0><<<dim3(256, 5), blk, 0, stream>>>(KX, wkf, bk, 192, 288, KPb, nullptr, nullptr);
    gemm_wmma<0,0><<<dim3(256, 5), blk, 0, stream>>>(KX + (size_t)192 * P, wkf, bk, 192, 288,
                                                     KPb + (size_t)288 * P, nullptr, nullptr);
    gemm_wmma<0,0><<<dim3(256, 5), blk, 0, stream>>>(VX, wvf, bvv, 192, 288, VPb, nullptr, nullptr);
    gemm_wmma<0,0><<<dim3(256, 5), blk, 0, stream>>>(VX + (size_t)192 * P, wvf, bvv, 192, 288,
                                                     VPb + (size_t)288 * P, nullptr, nullptr);

    // --- TDM prefetch of k/v planes ahead of the gather kernel ---
    k_tdm_prefetch<<<dim3(1), dim3(32), 0, stream>>>(KPb, 64, 4096);
    k_tdm_prefetch<<<dim3(1), dim3(32), 0, stream>>>(VPb, 64, 4096);

    // --- deformable attention (VALU gather + softmax) ---
    deform_attn<<<dim3(192), t256, 0, stream>>>(QPb, KPb, VPb, OFF, AT);

    // --- output projection + MLP residual (Kpad 320 / 192 / 320) ---
    gemm_wmma<0,4><<<dim3(256, 3), blk, 0, stream>>>(AT, wpf, bp, 320, 144, O16, O32, nullptr);
    gemm_wmma<0,2><<<dim3(256, 5), blk, 0, stream>>>(O16, wm1f, bm1, 192, 288, M1, nullptr, nullptr);
    gemm_wmma<0,5><<<dim3(256, 3), blk, 0, stream>>>(M1, wm2f, bm2, 320, 144, nullptr,
                                                     (float*)d_out, O32);
}
